// CrossAttention_Seg_41532333752472
// MI455X (gfx1250) — matrix operand order repaired and checked by reading, not compiled, not run on hardware
//
#include <hip/hip_runtime.h>
#include <hip/hip_bf16.h>

typedef __attribute__((ext_vector_type(16))) _Float16 v16h;
typedef __attribute__((ext_vector_type(8)))  _Float16 v8h;
typedef __attribute__((ext_vector_type(8)))  float    v8f;

#define B_  8
#define N_  2048
#define S_  1024
#define K_  32
#define P0_ (B_*N_)      // 16384 : per-point spatial size
#define P1_ (B_*S_*K_)   // 262144: grouped spatial size (b,k,s)
#define P2_ (B_*S_)      // 8192  : center spatial size (b,s)
#define BN_EPS_ 1e-5f

static __host__ __device__ inline int r32i(int x) { return (x + 31) & ~31; }

// ---------------------------------------------------------------------------
// Pack W (f32 [O,C]) -> Wp (f16 [OP, Cpad]) with zero padding: removes all
// bounds checks from the GEMM inner loop.
// ---------------------------------------------------------------------------
__global__ void pack_w_kernel(const float* __restrict__ W, _Float16* __restrict__ Wp,
                              int O, int C, int OP, int Cpad)
{
    int i = blockIdx.x * blockDim.x + threadIdx.x;
    if (i >= OP * Cpad) return;
    int o = i / Cpad, c = i % Cpad;
    Wp[i] = (o < O && c < C) ? (_Float16)W[o * C + c] : (_Float16)0.f;
}

// ---------------------------------------------------------------------------
// WMMA GEMM, activations point-major: Y[p][OP] = Wp[OP,Cpad] * X[p][Cpad]
// One wave: 16 points x all M tiles. All fragment loads are aligned b128.
// Fragment k order: contiguous k = 16*(lane/16) + e for both A and B (wave32).
// ---------------------------------------------------------------------------
template<int MT>
__global__ __launch_bounds__(128)
void gemm_wmma_kernel(const _Float16* __restrict__ Wp, const _Float16* __restrict__ X,
                      _Float16* __restrict__ Y, int Cpad, int P)
{
    constexpr int OP = MT * 16;
    const int lane = threadIdx.x & 31;
    const int wave = threadIdx.x >> 5;
    const int col  = (blockIdx.x * 4 + wave) * 16 + (lane & 15);
    const int lohi = lane >> 4;                  // 0: lanes 0-15, 1: lanes 16-31
    const _Float16* xrow = X + (size_t)col * Cpad;
    __builtin_prefetch(xrow, 0, 3);              // global_prefetch_b8

    v8f acc[MT];
#pragma unroll
    for (int t = 0; t < MT; ++t) acc[t] = v8f{0.f,0.f,0.f,0.f,0.f,0.f,0.f,0.f};

    const int a0off = lohi ? 16 : 0;             // A frag: K = kk+{a0off..a0off+7}
    const int a1off = lohi ? 24 : 8;             //         K = kk+{a1off..a1off+7}
    const int boff  = lohi ? 16 : 0;             // B frag: K = kk+boff+e
    const _Float16* wbase = Wp + (size_t)(lane & 15) * Cpad;

    for (int kk = 0; kk < Cpad; kk += 32) {
        v8h b0 = *(const v8h*)(xrow + kk + boff);
        v8h b1 = *(const v8h*)(xrow + kk + boff + 8);
        v16h bf = __builtin_shufflevector(b0, b1, 0,1,2,3,4,5,6,7,8,9,10,11,12,13,14,15);
#pragma unroll
        for (int t = 0; t < MT; ++t) {
            const _Float16* wr = wbase + (size_t)t * 16 * Cpad + kk;
            v8h a0 = *(const v8h*)(wr + a0off);
            v8h a1 = *(const v8h*)(wr + a1off);
            v16h af = __builtin_shufflevector(a0, a1, 0,1,2,3,4,5,6,7,8,9,10,11,12,13,14,15);
            acc[t] = __builtin_amdgcn_wmma_f32_16x16x32_f16(
                false, af, false, bf, (short)0, acc[t], false, false);
        }
    }
#pragma unroll
    for (int t = 0; t < MT; ++t) {
        v8h st;
#pragma unroll
        for (int r = 0; r < 8; ++r) st[r] = (_Float16)acc[t][r];
        *(v8h*)(Y + (size_t)col * OP + t * 16 + lohi * 8) = st;   // aligned b128 store
    }
}

// ---------------------------------------------------------------------------
// BN statistics, coalesced over [p][c]: thread owns one channel, strides points.
// blockDim = 2*OP (64..256). sums[2c]=sum, sums[2c+1]=sumsq (atomic f32).
// ---------------------------------------------------------------------------
__global__ void bn_stats_kernel(const _Float16* __restrict__ Y, float* __restrict__ sums,
                                int OP, int P)
{
    const int c = threadIdx.x % OP;
    const int r = threadIdx.x / OP;             // 0..1
    float s = 0.f, s2 = 0.f;
    for (int p = blockIdx.x * 2 + r; p < P; p += gridDim.x * 2) {
        float v = (float)Y[(size_t)p * OP + c];
        s += v; s2 += v * v;
    }
    atomicAdd(&sums[2 * c], s);
    atomicAdd(&sums[2 * c + 1], s2);
}

__global__ void bn_relu_kernel(_Float16* __restrict__ Y, const float* __restrict__ sums,
                               const float* __restrict__ g, const float* __restrict__ be,
                               int O, int OP, int P)
{
    size_t i = (size_t)blockIdx.x * blockDim.x + threadIdx.x;
    if (i >= (size_t)P * OP) return;
    int c = (int)(i % (size_t)OP);
    if (c >= O) { Y[i] = (_Float16)0.f; return; }
    float m = sums[2 * c] / (float)P;
    float v = sums[2 * c + 1] / (float)P - m * m;
    float x = ((float)Y[i] - m) * rsqrtf(v + BN_EPS_) * g[c] + be[c];
    Y[i] = (_Float16)fmaxf(x, 0.f);
}

// Final layer: BN+ReLU straight into f32 output; [p][128] == [B,N,128] flat.
__global__ void bn_relu_out_kernel(const _Float16* __restrict__ Y, const float* __restrict__ sums,
                                   const float* __restrict__ g, const float* __restrict__ be,
                                   float* __restrict__ out, int P)
{
    size_t i = (size_t)blockIdx.x * blockDim.x + threadIdx.x;
    if (i >= (size_t)P * 128) return;
    int c = (int)(i % 128u);
    float m = sums[2 * c] / (float)P;
    float v = sums[2 * c + 1] / (float)P - m * m;
    float x = ((float)Y[i] - m) * rsqrtf(v + BN_EPS_) * g[c] + be[c];
    out[i] = fmaxf(x, 0.f);
}

__global__ void zero_f32_kernel(float* __restrict__ y, int n)
{
    int i = blockIdx.x * blockDim.x + threadIdx.x;
    if (i < n) y[i] = 0.f;
}

// ---------------------------------------------------------------------------
// pts f16 [P0][32] (channels 12..31 zero)
// ---------------------------------------------------------------------------
__global__ void build_pts_kernel(const float* __restrict__ xyz, const float* __restrict__ eula,
                                 const float* __restrict__ nearp, const float* __restrict__ meta,
                                 _Float16* __restrict__ pts)
{
    int p = blockIdx.x * blockDim.x + threadIdx.x;
    if (p >= P0_) return;
    _Float16* o = pts + (size_t)p * 32;
    o[0] = (_Float16)xyz[p * 3 + 0];  o[1] = (_Float16)xyz[p * 3 + 1];  o[2] = (_Float16)xyz[p * 3 + 2];
    o[3] = (_Float16)eula[p * 3 + 0]; o[4] = (_Float16)eula[p * 3 + 1]; o[5] = (_Float16)eula[p * 3 + 2];
    o[6] = (_Float16)nearp[p * 2 + 0]; o[7] = (_Float16)nearp[p * 2 + 1];
#pragma unroll
    for (int d = 0; d < 4; ++d) o[8 + d] = (_Float16)meta[p * 4 + d];
#pragma unroll
    for (int d = 12; d < 32; ++d) o[d] = (_Float16)0.f;
}

// ---------------------------------------------------------------------------
// KNN: one wave per query point. xyz[b] staged to LDS via async copy
// (global_load_async_to_lds_b128, ASYNCcnt), then 32 rounds of wave argmin.
// ---------------------------------------------------------------------------
__global__ __launch_bounds__(128)
void knn_kernel(const float* __restrict__ xyz, int* __restrict__ knn)
{
    __shared__ float sxyz[N_ * 3];               // 24 KB
    __shared__ float dist[4][N_];                // 32 KB
    const int wave = threadIdx.x >> 5, lane = threadIdx.x & 31;
    const int pi = blockIdx.x * 4 + wave;        // b*N + i (4 points share one b)
    const int b = pi / N_, i = pi % N_;
    const float* gb = xyz + (size_t)b * N_ * 3;

    // async stage: 6144 floats = 1536 x b128
    for (int t = threadIdx.x; t < (N_ * 3) / 4; t += 128) {
        unsigned la = (unsigned)(size_t)(&sxyz[t * 4]);
        const float* gp = gb + t * 4;
        asm volatile("global_load_async_to_lds_b128 %0, %1, off"
                     :: "v"(la), "v"(gp) : "memory");
    }
    asm volatile("s_wait_asynccnt 0x0" ::: "memory");
    __syncthreads();

    const float xi = sxyz[i * 3 + 0], yi = sxyz[i * 3 + 1], zi = sxyz[i * 3 + 2];
    for (int j = lane; j < N_; j += 32) {
        float dx = sxyz[j * 3 + 0] - xi;
        float dy = sxyz[j * 3 + 1] - yi;
        float dz = sxyz[j * 3 + 2] - zi;
        dist[wave][j] = dx * dx + dy * dy + dz * dz;
    }
    __syncthreads();
    for (int k = 0; k < K_; ++k) {
        float best = 1e30f; int bidx = 0x7fffffff;
        for (int j = lane; j < N_; j += 32) {
            float d = dist[wave][j];
            if (d < best) { best = d; bidx = j; }
        }
        for (int off = 16; off; off >>= 1) {
            float ob = __shfl_xor(best, off, 32);
            int   oi = __shfl_xor(bidx, off, 32);
            if (ob < best || (ob == best && oi < bidx)) { best = ob; bidx = oi; }
        }
        if (lane == 0) {
            knn[(size_t)pi * K_ + k] = bidx;
            dist[wave][bidx] = 1e30f;
        }
        __syncthreads();
    }
}

// ---------------------------------------------------------------------------
// FPS: one block per batch; serial 1024-step scan, dist in LDS
// ---------------------------------------------------------------------------
__global__ __launch_bounds__(256)
void fps_kernel(const float* __restrict__ xyz, int* __restrict__ fpsi)
{
    const int b = blockIdx.x;
    const int tid = threadIdx.x;
    __shared__ float dist[N_];
    __shared__ float rv[256];
    __shared__ int   ri[256];
    __shared__ int   s_far;
    const float* base = xyz + (size_t)b * N_ * 3;
    for (int j = tid; j < N_; j += 256) dist[j] = 1e10f;
    if (tid == 0) s_far = 0;
    __syncthreads();
    for (int s = 0; s < S_; ++s) {
        const int far = s_far;
        if (tid == 0) fpsi[b * S_ + s] = far;
        const float cx = base[far * 3 + 0], cy = base[far * 3 + 1], cz = base[far * 3 + 2];
        float lbest = -1.f; int lidx = 0;
        for (int j = tid; j < N_; j += 256) {
            float dx = base[j * 3 + 0] - cx;
            float dy = base[j * 3 + 1] - cy;
            float dz = base[j * 3 + 2] - cz;
            float nd = fminf(dist[j], dx * dx + dy * dy + dz * dz);
            dist[j] = nd;
            if (nd > lbest) { lbest = nd; lidx = j; }
        }
        rv[tid] = lbest; ri[tid] = lidx;
        __syncthreads();
        for (int w = 128; w > 0; w >>= 1) {
            if (tid < w) {
                if (rv[tid + w] > rv[tid] ||
                    (rv[tid + w] == rv[tid] && ri[tid + w] < ri[tid])) {
                    rv[tid] = rv[tid + w]; ri[tid] = ri[tid + w];
                }
            }
            __syncthreads();
        }
        if (tid == 0) s_far = ri[0];
        __syncthreads();
    }
}

__global__ void cxyz_kernel(const float* __restrict__ xyz, const int* __restrict__ fpsi,
                            float* __restrict__ cxyz)
{
    int t = blockIdx.x * blockDim.x + threadIdx.x;
    if (t >= P2_) return;
    int b = t / S_;
    size_t pc = (size_t)b * N_ + fpsi[t];
    cxyz[t * 3 + 0] = xyz[pc * 3 + 0];
    cxyz[t * 3 + 1] = xyz[pc * 3 + 1];
    cxyz[t * 3 + 2] = xyz[pc * 3 + 2];
}

// ---------------------------------------------------------------------------
// g_fea f16 [P1][64] (50 real + 14 zero), p = (b*K + k)*S + s
// ---------------------------------------------------------------------------
__global__ void gather_group_kernel(const float* __restrict__ xyz, const float* __restrict__ eula,
                                    const float* __restrict__ nearp, const float* __restrict__ meta,
                                    const _Float16* __restrict__ fea0, const int* __restrict__ knn,
                                    const int* __restrict__ fpsi, _Float16* __restrict__ g)
{
    int p = blockIdx.x * blockDim.x + threadIdx.x;
    if (p >= P1_) return;
    int s = p % S_;
    int bk = p / S_;
    int k = bk % K_;
    int b = bk / K_;
    int c = fpsi[b * S_ + s];
    int j = knn[((size_t)b * N_ + c) * K_ + k];
    size_t pj = (size_t)b * N_ + j, pc = (size_t)b * N_ + c;
    _Float16* o = g + (size_t)p * 64;
    o[0] = (_Float16)(xyz[pj * 3 + 0] - xyz[pc * 3 + 0]);
    o[1] = (_Float16)(xyz[pj * 3 + 1] - xyz[pc * 3 + 1]);
    o[2] = (_Float16)(xyz[pj * 3 + 2] - xyz[pc * 3 + 2]);
    o[3] = (_Float16)(eula[pj * 3 + 0] - eula[pc * 3 + 0]);
    o[4] = (_Float16)(eula[pj * 3 + 1] - eula[pc * 3 + 1]);
    o[5] = (_Float16)(eula[pj * 3 + 2] - eula[pc * 3 + 2]);
    o[6] = (_Float16)nearp[pj * 2 + 0];
    o[7] = (_Float16)nearp[pj * 2 + 1];
    o[8] = (_Float16)nearp[pc * 2 + 0];
    o[9] = (_Float16)nearp[pc * 2 + 1];
#pragma unroll
    for (int d = 0; d < 4; ++d) {
        o[10 + d] = (_Float16)meta[pj * 4 + d];
        o[14 + d] = (_Float16)meta[pc * 4 + d];
    }
    const _Float16* f = fea0 + pj * 32;
#pragma unroll
    for (int ch = 0; ch < 32; ++ch) o[18 + ch] = f[ch];
#pragma unroll
    for (int ch = 50; ch < 64; ++ch) o[ch] = (_Float16)0.f;
}

// ---------------------------------------------------------------------------
// c_att f16 [P2][64] (50 real + 14 zero) and c_fea f16 [P2][32]
// ---------------------------------------------------------------------------
__global__ void build_catt_kernel(const float* __restrict__ nearp, const float* __restrict__ meta,
                                  const _Float16* __restrict__ fea0, const int* __restrict__ fpsi,
                                  _Float16* __restrict__ catt, _Float16* __restrict__ cfea)
{
    int t = blockIdx.x * blockDim.x + threadIdx.x;
    if (t >= P2_) return;
    int b = t / S_;
    size_t pc = (size_t)b * N_ + fpsi[t];
    _Float16* o = catt + (size_t)t * 64;
#pragma unroll
    for (int ch = 0; ch < 6; ++ch) o[ch] = (_Float16)0.f;
    o[6] = (_Float16)nearp[pc * 2 + 0];
    o[7] = (_Float16)nearp[pc * 2 + 1];
    o[8] = (_Float16)nearp[pc * 2 + 0];
    o[9] = (_Float16)nearp[pc * 2 + 1];
#pragma unroll
    for (int d = 0; d < 4; ++d) {
        o[10 + d] = (_Float16)meta[pc * 4 + d];
        o[14 + d] = (_Float16)meta[pc * 4 + d];
    }
    const _Float16* f = fea0 + pc * 32;
    _Float16* cf = cfea + (size_t)t * 32;
#pragma unroll
    for (int ch = 0; ch < 32; ++ch) { o[18 + ch] = f[ch]; cf[ch] = f[ch]; }
#pragma unroll
    for (int ch = 50; ch < 64; ++ch) o[ch] = (_Float16)0.f;
}

// fmp[p][c] = fai80[(b*S+s)][c] - psi80[p][c]   (pitch 96, pads are 0-0=0)
__global__ void sub_bcast_kernel(const _Float16* __restrict__ fai, const _Float16* __restrict__ psi,
                                 _Float16* __restrict__ fmp)
{
    size_t i = (size_t)blockIdx.x * blockDim.x + threadIdx.x;
    if (i >= (size_t)P1_ * 96) return;
    int c = (int)(i % 96u);
    int p = (int)(i / 96u);
    int s = p % S_;
    int b = p / (S_ * K_);
    fmp[i] = (_Float16)((float)fai[((size_t)b * S_ + s) * 96 + c] - (float)psi[i]);
}

// ---------------------------------------------------------------------------
// Attention combine: one wave per p1; channel softmax (80 of 96) + K-sum via
// coalesced f32 atomics into y [P2][80]
// ---------------------------------------------------------------------------
__global__ __launch_bounds__(256)
void attn_combine_kernel(const _Float16* __restrict__ gam, const _Float16* __restrict__ alpha,
                         float* __restrict__ y)
{
    const int wave = threadIdx.x >> 5, lane = threadIdx.x & 31;
    const int p = blockIdx.x * 8 + wave;
    if (p >= P1_) return;
    const int s = p % S_;
    const int b = p / (S_ * K_);
    const _Float16* gp = gam + (size_t)p * 96;
    const _Float16* ap = alpha + (size_t)p * 96;
    float vals[3];
    float m = -1e30f;
#pragma unroll
    for (int t = 0; t < 3; ++t) {
        int c = t * 32 + lane;
        vals[t] = (c < 80) ? (float)gp[c] : -1e30f;
        m = fmaxf(m, vals[t]);
    }
#pragma unroll
    for (int off = 16; off; off >>= 1) m = fmaxf(m, __shfl_xor(m, off, 32));
    float ev[3]; float se = 0.f;
#pragma unroll
    for (int t = 0; t < 3; ++t) {
        int c = t * 32 + lane;
        ev[t] = (c < 80) ? __expf(vals[t] - m) : 0.f;
        se += ev[t];
    }
#pragma unroll
    for (int off = 16; off; off >>= 1) se += __shfl_xor(se, off, 32);
    const float scale = (80.f / (float)K_) / se;
    float* yp = y + ((size_t)b * S_ + s) * 80;
#pragma unroll
    for (int t = 0; t < 3; ++t) {
        int c = t * 32 + lane;
        if (c < 80) atomicAdd(&yp[c], ev[t] * scale * (float)ap[c]);
    }
}

// sa_out f16 [P2][112] = concat(c_fea 32, y 80)
__global__ void concat_saout_kernel(const _Float16* __restrict__ cfea, const float* __restrict__ y,
                                    _Float16* __restrict__ sa)
{
    size_t i = (size_t)blockIdx.x * blockDim.x + threadIdx.x;
    if (i >= (size_t)P2_ * 112) return;
    int ch = (int)(i % 112u);
    int p = (int)(i / 112u);
    sa[i] = (ch < 32) ? cfea[(size_t)p * 32 + ch]
                      : (_Float16)y[(size_t)p * 80 + (ch - 32)];
}

// ---------------------------------------------------------------------------
// 3-NN inverse-distance interp + concat -> X144 f16 [P0][160] (144 real)
// one wave per (b,n); per-lane sorted top-3 merged via xor butterfly
// ---------------------------------------------------------------------------
__device__ __forceinline__ void ins3(float d, int i,
                                     float& d0, int& i0, float& d1, int& i1,
                                     float& d2, int& i2)
{
    if (d < d0)      { d2 = d1; i2 = i1; d1 = d0; i1 = i0; d0 = d; i0 = i; }
    else if (d < d1) { d2 = d1; i2 = i1; d1 = d;  i1 = i; }
    else if (d < d2) { d2 = d;  i2 = i; }
}

__global__ __launch_bounds__(256)
void fp_interp_kernel(const float* __restrict__ xyz, const float* __restrict__ cxyz,
                      const _Float16* __restrict__ fea0, const _Float16* __restrict__ sa,
                      _Float16* __restrict__ X144)
{
    const int wave = threadIdx.x >> 5, lane = threadIdx.x & 31;
    const int pi = blockIdx.x * 8 + wave;
    if (pi >= P0_) return;
    const int b = pi / N_;
    const float xi = xyz[pi * 3 + 0], yi = xyz[pi * 3 + 1], zi = xyz[pi * 3 + 2];
    const float* cb = cxyz + (size_t)b * S_ * 3;
    float d0 = 1e30f, d1 = 1e30f, d2 = 1e30f;
    int i0 = 0, i1 = 0, i2 = 0;
    for (int s = lane; s < S_; s += 32) {
        float dx = cb[s * 3 + 0] - xi;
        float dy = cb[s * 3 + 1] - yi;
        float dz = cb[s * 3 + 2] - zi;
        ins3(dx * dx + dy * dy + dz * dz, s, d0, i0, d1, i1, d2, i2);
    }
    for (int off = 16; off; off >>= 1) {
        float e0 = __shfl_xor(d0, off, 32), e1 = __shfl_xor(d1, off, 32), e2 = __shfl_xor(d2, off, 32);
        int   j0 = __shfl_xor(i0, off, 32), j1 = __shfl_xor(i1, off, 32), j2 = __shfl_xor(i2, off, 32);
        ins3(e0, j0, d0, i0, d1, i1, d2, i2);
        ins3(e1, j1, d0, i0, d1, i1, d2, i2);
        ins3(e2, j2, d0, i0, d1, i1, d2, i2);
    }
    float w0 = 1.f / (d0 + 1e-8f), w1 = 1.f / (d1 + 1e-8f), w2 = 1.f / (d2 + 1e-8f);
    float ws = w0 + w1 + w2;
    w0 /= ws; w1 /= ws; w2 /= ws;
    const _Float16* s0 = sa + ((size_t)b * S_ + i0) * 112;
    const _Float16* s1 = sa + ((size_t)b * S_ + i1) * 112;
    const _Float16* s2 = sa + ((size_t)b * S_ + i2) * 112;
    const _Float16* f0 = fea0 + (size_t)pi * 32;
    _Float16* o = X144 + (size_t)pi * 160;
    for (int ch = lane; ch < 160; ch += 32) {
        float v;
        if (ch < 32)       v = (float)f0[ch];
        else if (ch < 144) {
            int c = ch - 32;
            v = w0 * (float)s0[c] + w1 * (float)s1[c] + w2 * (float)s2[c];
        } else v = 0.f;
        o[ch] = (_Float16)v;
    }
}

// ---------------------------------------------------------------------------
// Host-side helpers
// ---------------------------------------------------------------------------
static void gemm(const _Float16* Wp, const _Float16* X, _Float16* Y,
                 int OP, int Cpad, int P, hipStream_t st)
{
    dim3 grid(P / 64), blk(128);
    switch (OP / 16) {
        case 2:  gemm_wmma_kernel<2><<<grid, blk, 0, st>>>(Wp, X, Y, Cpad, P); break;
        case 4:  gemm_wmma_kernel<4><<<grid, blk, 0, st>>>(Wp, X, Y, Cpad, P); break;
        case 6:  gemm_wmma_kernel<6><<<grid, blk, 0, st>>>(Wp, X, Y, Cpad, P); break;
        default: gemm_wmma_kernel<8><<<grid, blk, 0, st>>>(Wp, X, Y, Cpad, P); break;
    }
}

static void conv_bn_relu(const float* W, const float* g, const float* be,
                         const _Float16* X, _Float16* Y, int O, int C, int P,
                         _Float16* Wp, float* sums, hipStream_t st)
{
    const int Cpad = r32i(C), OP = r32i(O);
    pack_w_kernel<<<(OP * Cpad + 255) / 256, 256, 0, st>>>(W, Wp, O, C, OP, Cpad);
    gemm(Wp, X, Y, OP, Cpad, P, st);
    zero_f32_kernel<<<1, 256, 0, st>>>(sums, 2 * OP);
    bn_stats_kernel<<<512, 2 * OP, 0, st>>>(Y, sums, OP, P);
    size_t tot = (size_t)P * OP;
    bn_relu_kernel<<<(unsigned)((tot + 255) / 256), 256, 0, st>>>(Y, sums, g, be, O, OP, P);
}

extern "C" void kernel_launch(void* const* d_in, const int* in_sizes, int n_in,
                              void* d_out, int out_size, void* d_ws, size_t ws_size,
                              hipStream_t stream)
{
    (void)in_sizes; (void)n_in; (void)out_size; (void)ws_size;
    const float* xyz   = (const float*)d_in[0];
    const float* eula  = (const float*)d_in[1];
    const float* nearp = (const float*)d_in[2];
    const float* meta  = (const float*)d_in[3];
    const float *Wpre0 = (const float*)d_in[4],  *Gpre0 = (const float*)d_in[6],  *Bpre0 = (const float*)d_in[7];
    const float *Wpre1 = (const float*)d_in[8],  *Gpre1 = (const float*)d_in[10], *Bpre1 = (const float*)d_in[11];
    const float *Wsa0 = (const float*)d_in[12], *Gsa0 = (const float*)d_in[14], *Bsa0 = (const float*)d_in[15];
    const float *Wsa1 = (const float*)d_in[16], *Gsa1 = (const float*)d_in[18], *Bsa1 = (const float*)d_in[19];
    const float *Wsa2 = (const float*)d_in[20], *Gsa2 = (const float*)d_in[22], *Bsa2 = (const float*)d_in[23];
    const float *Wfa0 = (const float*)d_in[24], *Gfa0 = (const float*)d_in[26], *Bfa0 = (const float*)d_in[27];
    const float *Wfa1 = (const float*)d_in[28], *Gfa1 = (const float*)d_in[30], *Bfa1 = (const float*)d_in[31];
    const float *Wps0 = (const float*)d_in[32], *Gps0 = (const float*)d_in[34], *Bps0 = (const float*)d_in[35];
    const float *Wps1 = (const float*)d_in[36], *Gps1 = (const float*)d_in[38], *Bps1 = (const float*)d_in[39];
    const float *Wal0 = (const float*)d_in[40], *Gal0 = (const float*)d_in[42], *Bal0 = (const float*)d_in[43];
    const float *Wal1 = (const float*)d_in[44], *Gal1 = (const float*)d_in[46], *Bal1 = (const float*)d_in[47];
    const float *Wgm0 = (const float*)d_in[48], *Ggm0 = (const float*)d_in[50], *Bgm0 = (const float*)d_in[51];
    const float *Wgm1 = (const float*)d_in[52], *Ggm1 = (const float*)d_in[54], *Bgm1 = (const float*)d_in[55];
    const float *Wfp = (const float*)d_in[56], *Gfp = (const float*)d_in[58], *Bfp = (const float*)d_in[59];

    float* out = (float*)d_out;

    // workspace arena (deterministic)
    char* ws = (char*)d_ws;
    size_t off = 0;
    auto alloc = [&](size_t bytes) -> char* {
        char* p = ws + off;
        off += (bytes + 255) & ~(size_t)255;
        return p;
    };
    _Float16* ptsT  = (_Float16*)alloc((size_t)P0_ * 32 * 2);
    _Float16* h16   = (_Float16*)alloc((size_t)P0_ * 32 * 2);
    _Float16* fea0h = (_Float16*)alloc((size_t)P0_ * 32 * 2);
    int*      knn   = (int*)     alloc((size_t)P0_ * K_ * 4);
    int*      fpsi  = (int*)     alloc((size_t)P2_ * 4);
    float*    cxyz  = (float*)   alloc((size_t)P2_ * 3 * 4);
    _Float16* cbufA = (_Float16*)alloc((size_t)P2_ * 96 * 2);
    _Float16* cbufB = (_Float16*)alloc((size_t)P2_ * 96 * 2);
    _Float16* cbufC = (_Float16*)alloc((size_t)P2_ * 96 * 2);
    _Float16* cfeah = (_Float16*)alloc((size_t)P2_ * 32 * 2);
    _Float16* BIG0  = (_Float16*)alloc((size_t)P1_ * 96 * 2);
    _Float16* BIG1  = (_Float16*)alloc((size_t)P1_ * 96 * 2);
    _Float16* BIG2  = (_Float16*)alloc((size_t)P1_ * 96 * 2);
    _Float16* BIG3  = (_Float16*)alloc((size_t)P1_ * 96 * 2);
    float*    y32   = (float*)   alloc((size_t)P2_ * 80 * 4);
    _Float16* saout = (_Float16*)alloc((size_t)P2_ * 112 * 2);
    _Float16* X144  = (_Float16*)alloc((size_t)P0_ * 160 * 2);
    _Float16* Y128  = (_Float16*)alloc((size_t)P0_ * 128 * 2);
    _Float16* Wp    = (_Float16*)alloc((size_t)128 * 160 * 2);
    float*    sums  = (float*)   alloc((size_t)256 * 4);

    // 1. preprocess MLP 12->16->32 (bias cancels under training BN)
    build_pts_kernel<<<(P0_ + 255) / 256, 256, 0, stream>>>(xyz, eula, nearp, meta, ptsT);
    conv_bn_relu(Wpre0, Gpre0, Bpre0, ptsT, h16,   16, 12, P0_, Wp, sums, stream);
    conv_bn_relu(Wpre1, Gpre1, Bpre1, h16,  fea0h, 32, 16, P0_, Wp, sums, stream);

    // 2. KNN + FPS
    knn_kernel<<<P0_ / 4, 128, 0, stream>>>(xyz, knn);
    fps_kernel<<<B_, 256, 0, stream>>>(xyz, fpsi);
    cxyz_kernel<<<(P2_ + 255) / 256, 256, 0, stream>>>(xyz, fpsi, cxyz);

    // 3. grouping
    gather_group_kernel<<<(P1_ + 255) / 256, 256, 0, stream>>>(
        xyz, eula, nearp, meta, fea0h, knn, fpsi, BIG0);
    build_catt_kernel<<<(P2_ + 255) / 256, 256, 0, stream>>>(
        nearp, meta, fea0h, fpsi, cbufA, cfeah);

    // 4. SA MLP 50->64->72->80 (separate BN stats per tensor)
    conv_bn_relu(Wsa0, Gsa0, Bsa0, BIG0, BIG1, 64, 50, P1_, Wp, sums, stream);
    conv_bn_relu(Wsa1, Gsa1, Bsa1, BIG1, BIG2, 72, 64, P1_, Wp, sums, stream);
    conv_bn_relu(Wsa2, Gsa2, Bsa2, BIG2, BIG3, 80, 72, P1_, Wp, sums, stream);   // nf80
    conv_bn_relu(Wsa0, Gsa0, Bsa0, cbufA, cbufB, 64, 50, P2_, Wp, sums, stream);
    conv_bn_relu(Wsa1, Gsa1, Bsa1, cbufB, cbufA, 72, 64, P2_, Wp, sums, stream);
    conv_bn_relu(Wsa2, Gsa2, Bsa2, cbufA, cbufB, 80, 72, P2_, Wp, sums, stream); // c_att80

    // 5. attention MLPs 80->88->80
    conv_bn_relu(Wfa0, Gfa0, Bfa0, cbufB, cbufA, 88, 80, P2_, Wp, sums, stream);
    conv_bn_relu(Wfa1, Gfa1, Bfa1, cbufA, cbufC, 80, 88, P2_, Wp, sums, stream); // fai80
    conv_bn_relu(Wps0, Gps0, Bps0, BIG3, BIG0, 88, 80, P1_, Wp, sums, stream);
    conv_bn_relu(Wps1, Gps1, Bps1, BIG0, BIG1, 80, 88, P1_, Wp, sums, stream);   // psi80
    conv_bn_relu(Wal0, Gal0, Bal0, BIG3, BIG0, 88, 80, P1_, Wp, sums, stream);
    conv_bn_relu(Wal1, Gal1, Bal1, BIG0, BIG2, 80, 88, P1_, Wp, sums, stream);   // alpha80
    {
        size_t tot = (size_t)P1_ * 96;
        sub_bcast_kernel<<<(unsigned)((tot + 255) / 256), 256, 0, stream>>>(cbufC, BIG1, BIG0);
    }
    conv_bn_relu(Wgm0, Ggm0, Bgm0, BIG0, BIG1, 88, 80, P1_, Wp, sums, stream);
    conv_bn_relu(Wgm1, Ggm1, Bgm1, BIG1, BIG0, 80, 88, P1_, Wp, sums, stream);   // gam80

    // 6. softmax over channels, K-sum, scale by C/K
    zero_f32_kernel<<<(P2_ * 80 + 255) / 256, 256, 0, stream>>>(y32, P2_ * 80);
    attn_combine_kernel<<<P1_ / 8, 256, 0, stream>>>(BIG0, BIG2, y32);
    {
        size_t tot = (size_t)P2_ * 112;
        concat_saout_kernel<<<(unsigned)((tot + 255) / 256), 256, 0, stream>>>(cfeah, y32, saout);
    }

    // 7. feature propagation + final conv (output layout [p][128] == flat out)
    fp_interp_kernel<<<P0_ / 8, 256, 0, stream>>>(xyz, cxyz, fea0h, saout, X144);
    pack_w_kernel<<<(128 * 160 + 255) / 256, 256, 0, stream>>>(Wfp, Wp, 128, 144, 128, 160);
    gemm(Wp, X144, Y128, 128, 160, P0_, stream);
    zero_f32_kernel<<<1, 256, 0, stream>>>(sums, 256);
    bn_stats_kernel<<<512, 256, 0, stream>>>(Y128, sums, 128, P0_);
    {
        size_t tot = (size_t)P0_ * 128;
        bn_relu_out_kernel<<<(unsigned)((tot + 255) / 256), 256, 0, stream>>>(
            Y128, sums, Gfp, Bfp, out, P0_);
    }
}
